// HGPSLPool_2388001817203
// MI455X (gfx1250) — matrix operand order repaired and checked by reading, not compiled, not run on hardware
//
#include <hip/hip_runtime.h>
#include <hip/hip_bf16.h>
#include <math.h>

typedef __attribute__((ext_vector_type(16))) _Float16 v16h;
typedef __attribute__((ext_vector_type(8)))  float    v8f;

namespace {
constexpr int  cB    = 64;
constexpr int  cN    = 1024;
constexpr int  cD    = 128;
constexpr int  cNTOT = cB * cN;        // 65536
constexpr int  cETOT = cB * 8192;      // 524288
constexpr int  cK    = 820;            // kept nodes per graph
constexpr int  cBK   = cB * cK;        // 52480
constexpr int  cKK   = cK * cK;        // 672400
constexpr int  cWTOT = cB * cKK;       // 43,033,600 (fits int32)
constexpr float cNEG  = 0.2f;
constexpr float cLAMB = 1.0f;

__device__ __forceinline__ float lrelu(float x) { return x > 0.f ? x : cNEG * x; }

// ---------------------------------------------------------------------------
// 0) init: zero agg/out_deg/in_deg (contiguous 8,519,680 floats), maskarr=-1
// ---------------------------------------------------------------------------
__global__ __launch_bounds__(256) void k_init(float* zero_region, int* maskarr) {
    int t = blockIdx.x * 256 + threadIdx.x;          // grid covers 8,519,680
    zero_region[t] = 0.f;
    if (t < cNTOT) maskarr[t] = -1;
}

// ---------------------------------------------------------------------------
// 1) degrees (include self loops, as in reference)
// ---------------------------------------------------------------------------
__global__ __launch_bounds__(256) void k_degree(const int* src, const int* dst,
                                                float* out_deg, float* in_deg) {
    int e = blockIdx.x * 256 + threadIdx.x;          // exactly cETOT threads
    atomicAdd(&out_deg[src[e]], 1.f);
    atomicAdd(&in_deg[dst[e]], 1.f);
}

// ---------------------------------------------------------------------------
// 2) message scatter: agg[dst] += feat[src] * rsqrt(max(out_deg[src],1)) * e
//    one wave per edge -> coalesced 128-float row, skip self loops
// ---------------------------------------------------------------------------
__global__ __launch_bounds__(256) void k_scatter(const float* feat, const float* efeat,
                                                 const int* src, const int* dst,
                                                 const float* out_deg, float* agg) {
    int e    = blockIdx.x * 8 + (threadIdx.x >> 5); // 8 waves/block, exact cover
    int lane = threadIdx.x & 31;
    int s = src[e], d = dst[e];
    if (s == d) return;                              // remove_self_loop
    float coef = efeat[e] * rsqrtf(fmaxf(out_deg[s], 1.f));
    const float* frow = feat + (size_t)s * cD;
    float*       arow = agg  + (size_t)d * cD;
#pragma unroll
    for (int t = lane; t < cD; t += 32)
        atomicAdd(&arow[t], frow[t] * coef);
}

// ---------------------------------------------------------------------------
// 3) gate = sigmoid(leaky_relu((feat@W)@a)) — WMMA f32<-f16 16x16x32.
//    One wave computes a 16x128 tile of feat@W in 8 accumulators, contracts
//    with `a` in registers, butterfly-reduces across lanes.
// ---------------------------------------------------------------------------
__global__ __launch_bounds__(32) void k_gate_wmma(const float* __restrict__ feat,
                                                  const float* __restrict__ W,
                                                  const float* __restrict__ avec,
                                                  float* __restrict__ gate) {
    const int m0   = blockIdx.x * 16;
    const int lane = threadIdx.x;
    const int half = lane >> 4;       // lane group 0 / 1
    const int lm   = lane & 15;       // row (A) / col (B) within tile

    v8f acc[8];
    v8f zero = {0.f, 0.f, 0.f, 0.f, 0.f, 0.f, 0.f, 0.f};
#pragma unroll
    for (int t = 0; t < 8; ++t) acc[t] = zero;

    const float* arow = feat + (size_t)(m0 + lm) * cD;
#pragma unroll
    for (int kc = 0; kc < cD; kc += 32) {
        // A tile 16x32 f16: lane holds row lm; element e -> K per ISA layout
        v16h a;
#pragma unroll
        for (int e = 0; e < 16; ++e) {
            int kk = (e & 7) + 16 * (e >> 3) + 8 * half;
            a[e] = (_Float16)arow[kc + kk];
        }
#pragma unroll
        for (int t = 0; t < 8; ++t) {
            int n0 = t * 16;
            v16h b;                    // B tile 32x16: lane holds col lm
#pragma unroll
            for (int e = 0; e < 16; ++e) {
                int kk = (e & 7) + 16 * (e >> 3) + 8 * half;
                b[e] = (_Float16)W[(size_t)(kc + kk) * cD + n0 + lm];
            }
            acc[t] = __builtin_amdgcn_wmma_f32_16x16x32_f16(
                false, a, false, b, (short)0, acc[t], false, false);
        }
    }
    // contract with `a` vector: lane owns column N=lm; acc[t][v] is row v+8*half
    float part[8];
#pragma unroll
    for (int v = 0; v < 8; ++v) {
        float p = 0.f;
#pragma unroll
        for (int t = 0; t < 8; ++t) p += acc[t][v] * avec[t * 16 + lm];
        part[v] = p;
    }
#pragma unroll
    for (int off = 1; off < 16; off <<= 1)
#pragma unroll
        for (int v = 0; v < 8; ++v) part[v] += __shfl_xor(part[v], off, 32);
    if (lm == 0) {
#pragma unroll
        for (int v = 0; v < 8; ++v) {
            float s = lrelu(part[v]);
            gate[m0 + half * 8 + v] = 1.f / (1.f + __expf(-s));
        }
    }
}

// ---------------------------------------------------------------------------
// 4) x_score[n] = gate[n] * sum_d |feat[n,d] - agg[n,d]*dn[n]|  (wave/node)
// ---------------------------------------------------------------------------
__global__ __launch_bounds__(256) void k_info(const float* feat, const float* agg,
                                              const float* in_deg, const float* gate,
                                              float* xscore) {
    int node = blockIdx.x * 8 + (threadIdx.x >> 5);  // exact cover of cNTOT
    int lane = threadIdx.x & 31;
    float dn = rsqrtf(fmaxf(in_deg[node], 1.f));
    const float* f = feat + (size_t)node * cD;
    const float* g = agg  + (size_t)node * cD;
    float s = 0.f;
#pragma unroll
    for (int d = lane; d < cD; d += 32) s += fabsf(f[d] - g[d] * dn);
#pragma unroll
    for (int off = 1; off < 32; off <<= 1) s += __shfl_xor(s, off, 32);
    if (lane == 0) xscore[node] = s * gate[node];
}

// ---------------------------------------------------------------------------
// 5) per-graph top-K: bitonic sort of 1024 (value,index) pairs in LDS,
//    descending by value with ascending-index tie-break (matches lax.top_k).
//    Writes perm (float to d_out, int to ws) and maskarr[global]=pooled idx.
// ---------------------------------------------------------------------------
__global__ __launch_bounds__(256) void k_topk(const float* xscore, float* perm_f,
                                              int* perm_i, int* maskarr) {
    __shared__ float sv[cN];
    __shared__ int   si[cN];
    const int b = blockIdx.x, tid = threadIdx.x;
    for (int i = tid; i < cN; i += 256) { sv[i] = xscore[b * cN + i]; si[i] = i; }
    __syncthreads();
    for (int kk = 2; kk <= cN; kk <<= 1) {
        for (int jj = kk >> 1; jj > 0; jj >>= 1) {
            for (int i = tid; i < cN; i += 256) {
                int ixj = i ^ jj;
                if (ixj > i) {
                    float va = sv[i], vb = sv[ixj];
                    int   ia = si[i], ib = si[ixj];
                    bool before = (va > vb) || (va == vb && ia < ib);
                    bool dir = ((i & kk) == 0);
                    if (before != dir) {
                        sv[i] = vb; sv[ixj] = va; si[i] = ib; si[ixj] = ia;
                    }
                }
            }
            __syncthreads();
        }
    }
    for (int j = tid; j < cK; j += 256) {
        int g = b * cN + si[j];
        perm_i[b * cK + j] = g;
        perm_f[b * cK + j] = (float)g;
        maskarr[g] = b * cK + j;
    }
}

// ---------------------------------------------------------------------------
// 6) gather pooled features + wsrc/wdst matvecs (wave per pooled node)
// ---------------------------------------------------------------------------
__global__ __launch_bounds__(256) void k_gather(const float* feat, const int* perm_i,
                                                const float* att, float* feat_p,
                                                float* wsrc, float* wdst) {
    int i    = blockIdx.x * 8 + (threadIdx.x >> 5);  // exact cover of cBK
    int lane = threadIdx.x & 31;
    const float* row = feat + (size_t)perm_i[i] * cD;
    float s1 = 0.f, s2 = 0.f;
#pragma unroll
    for (int d = lane; d < cD; d += 32) {
        float v = row[d];
        feat_p[(size_t)i * cD + d] = v;
        s1 += v * att[d];
        s2 += v * att[cD + d];
    }
#pragma unroll
    for (int off = 1; off < 32; off <<= 1) {
        s1 += __shfl_xor(s1, off, 32);
        s2 += __shfl_xor(s2, off, 32);
    }
    if (lane == 0) { wsrc[i] = s1; wdst[i] = s2; }
}

// ---------------------------------------------------------------------------
// 7) Wt[b,j,i] = leaky_relu(wsrc[b,i] + wdst[b,j])  (dst-major / transposed)
// ---------------------------------------------------------------------------
__global__ __launch_bounds__(256) void k_fill(const float* wsrc, const float* wdst,
                                              float* Wt) {
    int idx = blockIdx.x * 256 + threadIdx.x;        // exact cover of cWTOT
    int b = idx / cKK, r = idx % cKK;
    int j = r / cK, i = r % cK;                      // i fast -> coalesced
    Wt[idx] = lrelu(wsrc[b * cK + i] + wdst[b * cK + j]);
}

// ---------------------------------------------------------------------------
// 8) add lambda*e_feat at pooled edges (transposed position)
// ---------------------------------------------------------------------------
__global__ __launch_bounds__(256) void k_edge_add(const int* src, const int* dst,
                                                  const float* efeat, const int* maskarr,
                                                  float* Wt) {
    int e = blockIdx.x * 256 + threadIdx.x;          // exactly cETOT
    int nr = maskarr[src[e]], nc = maskarr[dst[e]];
    if (nr >= 0 && nc >= 0) {
        int b = nr / cK, i = nr % cK, j = nc % cK;
        atomicAdd(&Wt[((size_t)(b * cK + j)) * cK + i], cLAMB * efeat[e]);
    }
}

// ---------------------------------------------------------------------------
// 9) sparsemax over each contiguous row of Wt (one block per (b,dst) column
//    of the original layout): bitonic sort desc + inclusive scan + tau.
// ---------------------------------------------------------------------------
__global__ __launch_bounds__(256) void k_sparsemax(float* Wt) {
    __shared__ float sh[cN];
    __shared__ float cs[cN];
    __shared__ int   cnt;
    __shared__ float tauS;
    const int row = blockIdx.x;                      // 0 .. cBK-1
    const int tid = threadIdx.x;
    float* z = Wt + (size_t)row * cK;

    for (int i = tid; i < cN; i += 256) sh[i] = (i < cK) ? z[i] : -3.0e38f;
    if (tid == 0) cnt = 0;
    __syncthreads();
    // bitonic sort descending (values only)
    for (int kk = 2; kk <= cN; kk <<= 1) {
        for (int jj = kk >> 1; jj > 0; jj >>= 1) {
            for (int i = tid; i < cN; i += 256) {
                int ixj = i ^ jj;
                if (ixj > i) {
                    float a = sh[i], b = sh[ixj];
                    bool dir = ((i & kk) == 0);
                    if ((a > b) != dir) { sh[i] = b; sh[ixj] = a; }
                }
            }
            __syncthreads();
        }
    }
    // inclusive scan (Hillis-Steele, 4 elems/thread)
    for (int i = tid; i < cN; i += 256) cs[i] = sh[i];
    __syncthreads();
    for (int off = 1; off < cN; off <<= 1) {
        float t[4];
#pragma unroll
        for (int u = 0; u < 4; ++u) {
            int i = tid + u * 256;
            t[u] = (i >= off) ? cs[i - off] : 0.f;
        }
        __syncthreads();
#pragma unroll
        for (int u = 0; u < 4; ++u) cs[tid + u * 256] += t[u];
        __syncthreads();
    }
    // support size: #{ r in 1..K : 1 + r*zs[r-1] > cs[r-1] }  (prefix property)
    int local = 0;
    for (int i = tid; i < cK; i += 256)
        if (1.0f + (float)(i + 1) * sh[i] > cs[i]) local++;
    atomicAdd(&cnt, local);
    __syncthreads();
    if (tid == 0) tauS = (cs[cnt - 1] - 1.0f) / (float)cnt;
    __syncthreads();
    float tau = tauS;
    for (int i = tid; i < cK; i += 256) {
        float w = fmaxf(z[i] - tau, 0.f);
        z[i] = (fabsf(w) > 1e-9f) ? w : 0.f;
    }
}

// ---------------------------------------------------------------------------
// 10) in-place per-graph transpose: weights[b,i,j] = Wt[b,j,i]
//     (172 MB ~ resident in 192 MB L2, so the strided side is cheap)
// ---------------------------------------------------------------------------
__global__ __launch_bounds__(256) void k_transpose(float* Wt) {
    int idx = blockIdx.x * 256 + threadIdx.x;
    int b = idx / cKK, r = idx % cKK;
    int i = r / cK, j = r % cK;
    if (j > i) {
        size_t p = ((size_t)(b * cK + i)) * cK + j;
        size_t q = ((size_t)(b * cK + j)) * cK + i;
        float a = Wt[p], c = Wt[q];
        Wt[p] = c; Wt[q] = a;
    }
}

} // namespace

extern "C" void kernel_launch(void* const* d_in, const int* in_sizes, int n_in,
                              void* d_out, int out_size, void* d_ws, size_t ws_size,
                              hipStream_t stream) {
    const float* feat  = (const float*)d_in[0];   // [65536,128]
    const float* efeat = (const float*)d_in[1];   // [524288]
    const float* W     = (const float*)d_in[2];   // [128,128]
    const float* avec  = (const float*)d_in[3];   // [128,1]
    const float* att   = (const float*)d_in[4];   // [1,256]
    const int*   src   = (const int*)d_in[5];     // [524288]
    const int*   dst   = (const int*)d_in[6];     // [524288]

    // d_out layout: feat_p | weights | perm | x_score
    float* out_fp   = (float*)d_out;                  // 6,717,440
    float* out_w    = out_fp + (size_t)cBK * cD;      // 43,033,600 (also scratch)
    float* out_perm = out_w + (size_t)cWTOT;          // 52,480
    float* out_xs   = out_perm + cBK;                 // 65,536

    // workspace layout (floats then ints), ~34 MB total
    float* agg     = (float*)d_ws;                    // 8,388,608
    float* out_deg = agg + (size_t)cNTOT * cD;        // 65,536
    float* in_deg  = out_deg + cNTOT;                 // 65,536
    float* gate    = in_deg + cNTOT;                  // 65,536
    float* wsrc    = gate + cNTOT;                    // 52,480
    float* wdst    = wsrc + cBK;                      // 52,480
    int*   maskarr = (int*)(wdst + cBK);              // 65,536
    int*   perm_i  = maskarr + cNTOT;                 // 52,480

    const int zeroN = cNTOT * cD + 2 * cNTOT;         // agg + both degrees
    k_init     <<<zeroN / 256, 256, 0, stream>>>(agg, maskarr);
    k_degree   <<<cETOT / 256, 256, 0, stream>>>(src, dst, out_deg, in_deg);
    k_scatter  <<<cETOT / 8,   256, 0, stream>>>(feat, efeat, src, dst, out_deg, agg);
    k_gate_wmma<<<cNTOT / 16,   32, 0, stream>>>(feat, W, avec, gate);
    k_info     <<<cNTOT / 8,   256, 0, stream>>>(feat, agg, in_deg, gate, out_xs);
    k_topk     <<<cB,          256, 0, stream>>>(out_xs, out_perm, perm_i, maskarr);
    k_gather   <<<cBK / 8,     256, 0, stream>>>(feat, perm_i, att, out_fp, wsrc, wdst);
    k_fill     <<<cWTOT / 256, 256, 0, stream>>>(wsrc, wdst, out_w);
    k_edge_add <<<cETOT / 256, 256, 0, stream>>>(src, dst, efeat, maskarr, out_w);
    k_sparsemax<<<cBK,         256, 0, stream>>>(out_w);
    k_transpose<<<cWTOT / 256, 256, 0, stream>>>(out_w);
}
